// Attention_549755814457
// MI455X (gfx1250) — matrix operand order repaired and checked by reading, not compiled, not run on hardware
//
#include <hip/hip_runtime.h>
#include <hip/hip_bf16.h>
#include <math.h>

// ---------------------------------------------------------------------------
// Fixed problem shape: B=4, Lq=512, H=32, D=128, hidden=4096, hist=1536,
// Lkv=2048, block_size=64, nblk=32.
// ---------------------------------------------------------------------------
#define B_    4
#define LQ_   512
#define H_    32
#define D_    128
#define HID_  4096
#define HIST_ 1536
#define LKV_  2048
#define BS_   64
#define NBLK_ 32

typedef __attribute__((ext_vector_type(16))) __bf16 v16bf;
typedef __attribute__((ext_vector_type(8)))  float  v8f;
// GCC-vector types to match the builtin declarations (clang-23 lane).
typedef unsigned int u32x4 __attribute__((vector_size(16)));
typedef int          i32x4 __attribute__((vector_size(16)));
typedef int          i32x8 __attribute__((vector_size(32)));
typedef int          v4i   __attribute__((vector_size(16)));

// CDNA5 async / TDM paths, guarded so the file compiles on either toolchain.
#if defined(__has_builtin)
#if __has_builtin(__builtin_amdgcn_global_load_async_to_lds_b128) && \
    __has_builtin(__builtin_amdgcn_s_wait_asynccnt)
#define USE_ASYNC 1
#else
#define USE_ASYNC 0
#endif
#if __has_builtin(__builtin_amdgcn_tensor_load_to_lds) && \
    __has_builtin(__builtin_amdgcn_s_wait_tensorcnt)
#define USE_TDM 1
#else
#define USE_TDM 0
#endif
#else
#define USE_ASYNC 0
#define USE_TDM 0
#endif

// Async-to-LDS pointer casts: global (AS1) source, LDS (AS3) destination,
// pointee type v4i to match the builtin signature.
#define GPTR(p) ((__attribute__((address_space(1))) v4i*)(p))
#define LPTR(p) ((__attribute__((address_space(3))) v4i*)(p))
#define LOFF(p) ((unsigned)(uintptr_t)((__attribute__((address_space(3))) void*)(p)))

union FragU { v16bf v; uint4 u[2]; };

// Assemble a 16x-bf16 WMMA fragment from two 16B reads (LDS or global).
static __device__ __forceinline__ v16bf mem_frag(const char* base, int off0, int off1) {
  FragU f;
  f.u[0] = *reinterpret_cast<const uint4*>(base + off0);
  f.u[1] = *reinterpret_cast<const uint4*>(base + off1);
  return f.v;
}

static __device__ __forceinline__ v8f wmma_bf16(v16bf a, v16bf b, v8f c) {
  return __builtin_amdgcn_wmma_f32_16x16x32_bf16(false, a, false, b, (short)0, c,
                                                 false, false);
}

#if USE_TDM
// Tensor Data Mover: 2D tile load (tile_d0 x tile_d1 elements of 2 bytes)
// from a row-major tensor (row stride = stride0 elements) into LDS, packed
// contiguously (row-major, tile_d0*2 bytes per row). D# layout per ISA
// CDNA5 ISA. 6-arg builtin (clang-23).
static __device__ __forceinline__ void tdm_load_2d_bf16(
    unsigned lds_addr, const void* tile_start, unsigned tensor_d0,
    unsigned tensor_d1, unsigned tile_d0, unsigned tile_d1,
    unsigned long long stride0) {
  unsigned long long ga = (unsigned long long)(uintptr_t)tile_start;
  u32x4 g0;
  g0[0] = 1u;                                   // count=1, user mode
  g0[1] = lds_addr;                             // lds_addr (bytes)
  g0[2] = (unsigned)(ga & 0xFFFFFFFFu);         // global_addr[31:0]
  g0[3] = (unsigned)((ga >> 32) & 0x01FFFFFFu)  // global_addr[56:32]
          | 0x80000000u;                        // type=2 ("image")
  i32x8 g1;
  g1[0] = (int)(1u << 16);                      // wg_mask=0, data_size=1 (2B)
  g1[1] = (int)((tensor_d0 & 0xFFFFu) << 16);   // tensor_dim0[15:0] @bit48
  g1[2] = (int)((tensor_d0 >> 16) | ((tensor_d1 & 0xFFFFu) << 16));
  g1[3] = (int)((tensor_d1 >> 16) | (tile_d0 << 16));   // tile_dim0 @bit112
  g1[4] = (int)(tile_d1 & 0xFFFFu);             // tile_dim1; tile_dim2=0
  g1[5] = (int)(unsigned)(stride0 & 0xFFFFFFFFu);       // dim0_stride lo
  g1[6] = (int)(unsigned)((stride0 >> 32) & 0xFFFFu);   // dim0_stride hi
  g1[7] = 0;
  i32x4 z4 = {0, 0, 0, 0};
  i32x8 z8 = {0, 0, 0, 0, 0, 0, 0, 0};
  __builtin_amdgcn_tensor_load_to_lds(g0, g1, z4, z4, z8, 0);
}
#endif

// ---------------------------------------------------------------------------
// fp32 -> bf16 elementwise convert (vectorized float4 loads).
// ---------------------------------------------------------------------------
__global__ void cvt_f32_bf16(const float4* __restrict__ in,
                             __bf16* __restrict__ out, int n4) {
  int i = blockIdx.x * blockDim.x + threadIdx.x;
  if (i >= n4) return;
  float4 v = in[i];
  int o = i * 4;
  out[o + 0] = (__bf16)v.x;
  out[o + 1] = (__bf16)v.y;
  out[o + 2] = (__bf16)v.z;
  out[o + 3] = (__bf16)v.w;
}

// ---------------------------------------------------------------------------
// GEMM: C[M,N] = A[M,K] @ B[N,K]^T, bf16 in, fp32 accumulate.
// Block tile 128(M) x 128(N), K-step 32, 8 waves as 4(M) x 2(N):
// each wave owns 32x64 = 8 WMMA tiles -> 8 WMMAs per 6 fragment loads.
// Double-buffered TDM pipeline: wave 0 issues the DMAs for step i+1 before
// waiting on step i (in-order TENSORcnt -> s_wait_tensorcnt(2) retires the
// older pair), so the tile DMA overlaps the WMMAs of the previous step.
// ---------------------------------------------------------------------------
__global__ __launch_bounds__(256) void gemm_bf16_wmma(
    const __bf16* __restrict__ A, const __bf16* __restrict__ Bw,
    void* __restrict__ Cout, int M, int N, int K, int out_is_bf16) {
  __shared__ __align__(16) __bf16 sA[2][128 * 32];  // [m][k], 64B rows
  __shared__ __align__(16) __bf16 sB[2][128 * 32];  // [n][k], 64B rows

  const int tid = threadIdx.x;
  const int wid = tid >> 5;
  const int lane = tid & 31;
  const int lh = lane & 15;
  const bool hihalf = lane >= 16;
  const int wm = wid & 3;        // M quadrant (32 rows)
  const int wn = wid >> 2;       // N half (64 cols)
  const int m0 = blockIdx.y * 128;
  const int n0 = blockIdx.x * 128;

  v8f acc[2][4] = {};

#if USE_TDM
  // Prologue: issue buffer-0 tile DMAs (A + B).
  if (tid < 32) {
    tdm_load_2d_bf16(LOFF(&sA[0][0]), A + (size_t)m0 * K, (unsigned)K,
                     (unsigned)M, 32u, 128u, (unsigned long long)K);
    tdm_load_2d_bf16(LOFF(&sB[0][0]), Bw + (size_t)n0 * K, (unsigned)K,
                     (unsigned)N, 32u, 128u, (unsigned long long)K);
  }
#else
  const uint4* gA = reinterpret_cast<const uint4*>(A);
  const uint4* gB = reinterpret_cast<const uint4*>(Bw);
  // Prologue: stage buffer 0 synchronously.
#pragma unroll
  for (int c = 0; c < 2; ++c) {
    int ch = tid + c * 256;
    int r = ch >> 2;
    int k8 = (ch & 3) * 8;
    *reinterpret_cast<uint4*>(&sA[0][r * 32 + k8]) =
        gA[((size_t)(m0 + r) * K + (size_t)k8) >> 3];
    *reinterpret_cast<uint4*>(&sB[0][r * 32 + k8]) =
        gB[((size_t)(n0 + r) * K + (size_t)k8) >> 3];
  }
  __syncthreads();
#endif

  int cur = 0;
  for (int k0 = 0; k0 < K; k0 += 32) {
    const int nxt = cur ^ 1;
    const bool hasnext = (k0 + 32 < K);
#if USE_TDM
    if (tid < 32) {
      if (hasnext) {
        tdm_load_2d_bf16(LOFF(&sA[nxt][0]), A + (size_t)m0 * K + k0 + 32,
                         (unsigned)K, (unsigned)M, 32u, 128u,
                         (unsigned long long)K);
        tdm_load_2d_bf16(LOFF(&sB[nxt][0]), Bw + (size_t)n0 * K + k0 + 32,
                         (unsigned)K, (unsigned)N, 32u, 128u,
                         (unsigned long long)K);
        __builtin_amdgcn_s_wait_tensorcnt(2);  // cur pair retired (in-order)
      } else {
        __builtin_amdgcn_s_wait_tensorcnt(0);
      }
    }
    __syncthreads();  // cur buffer ready for all waves
#else
    if (hasnext) {
#pragma unroll
      for (int c = 0; c < 2; ++c) {
        int ch = tid + c * 256;
        int r = ch >> 2;
        int k8 = (ch & 3) * 8;
        *reinterpret_cast<uint4*>(&sA[nxt][r * 32 + k8]) =
            gA[((size_t)(m0 + r) * K + (size_t)(k0 + 32 + k8)) >> 3];
        *reinterpret_cast<uint4*>(&sB[nxt][r * 32 + k8]) =
            gB[((size_t)(n0 + r) * K + (size_t)(k0 + 32 + k8)) >> 3];
      }
    }
#endif

    const char* cA = reinterpret_cast<const char*>(sA[cur]);
    const char* cB = reinterpret_cast<const char*>(sB[cur]);
    v16bf af[2], bfr[4];
#pragma unroll
    for (int i = 0; i < 2; ++i) {
      // A fragment (ISA 16-bit A layout: lanes 0-15 -> K {0..7,16..23}).
      int baseA = (wm * 32 + i * 16 + lh) * 64 + (hihalf ? 16 : 0);
      af[i] = mem_frag(cA, baseA, baseA + 32);
    }
#pragma unroll
    for (int j = 0; j < 4; ++j) {
      // B fragment: lane -> column n, same K layout as A ({0..7,16..23} / {8..15,24..31}).
      int baseB = (wn * 64 + j * 16 + lh) * 64 + (hihalf ? 16 : 0);
      bfr[j] = mem_frag(cB, baseB, baseB + 32);
    }
#pragma unroll
    for (int i = 0; i < 2; ++i)
#pragma unroll
      for (int j = 0; j < 4; ++j)
        acc[i][j] = wmma_bf16(af[i], bfr[j], acc[i][j]);

    __syncthreads();  // cur consumed; safe for next DMA / staged data visible
    cur = nxt;
  }

  // Epilogue. C layout: VGPR r -> M = r (+8 for lanes 16-31), lane%16 -> N.
#pragma unroll
  for (int i = 0; i < 2; ++i) {
#pragma unroll
    for (int j = 0; j < 4; ++j) {
#pragma unroll
      for (int r = 0; r < 8; ++r) {
        int m = m0 + wm * 32 + i * 16 + r + (hihalf ? 8 : 0);
        int n = n0 + wn * 64 + j * 16 + lh;
        float v = acc[i][j][r];
        if (out_is_bf16)
          reinterpret_cast<__bf16*>(Cout)[(size_t)m * N + n] = (__bf16)v;
        else
          reinterpret_cast<float*>(Cout)[(size_t)m * N + n] = v;
      }
    }
  }
}

// ---------------------------------------------------------------------------
// RoPE + QKV split + relayout. One thread per (b,q,h,j), j in [0,64):
// handles the rotation pair (j, j+64). Writes Q[B][H][Lq][D],
// K/V[B][H][pos][D] at pos = hist + q.
// ---------------------------------------------------------------------------
__global__ void rope_split_kernel(const __bf16* __restrict__ qkv,
                                  __bf16* __restrict__ Qb,
                                  __bf16* __restrict__ Kb,
                                  __bf16* __restrict__ Vb) {
  int t = blockIdx.x * blockDim.x + threadIdx.x;  // B*LQ*H*64 threads
  int j = t & 63;
  int h = (t >> 6) & (H_ - 1);
  int q = (t >> 11) & (LQ_ - 1);
  int b = t >> 20;

  size_t row = (size_t)(b * LQ_ + q) * (3 * HID_);
  int col = h * D_ + j;
  float q0 = (float)qkv[row + col];
  float q1 = (float)qkv[row + col + 64];
  float k0 = (float)qkv[row + HID_ + col];
  float k1 = (float)qkv[row + HID_ + col + 64];
  float v0 = (float)qkv[row + 2 * HID_ + col];
  float v1 = (float)qkv[row + 2 * HID_ + col + 64];

  float pos = (float)(HIST_ + q);
  float inv = __powf(10000.0f, -((float)(2 * j)) / 128.0f);
  float ang = pos * inv;
  float s, c;
  __sincosf(ang, &s, &c);

  float qo0 = q0 * c - q1 * s, qo1 = q1 * c + q0 * s;
  float ko0 = k0 * c - k1 * s, ko1 = k1 * c + k0 * s;

  size_t qdst = ((size_t)(b * H_ + h) * LQ_ + q) * D_ + j;
  Qb[qdst] = (__bf16)qo0;
  Qb[qdst + 64] = (__bf16)qo1;
  size_t kdst = ((size_t)(b * H_ + h) * LKV_ + (HIST_ + q)) * D_ + j;
  Kb[kdst] = (__bf16)ko0;
  Kb[kdst + 64] = (__bf16)ko1;
  Vb[kdst] = (__bf16)v0;
  Vb[kdst + 64] = (__bf16)v1;
}

// ---------------------------------------------------------------------------
// Gather paged history K/V (fp32) into contiguous bf16 [B][H][pos][D].
// ---------------------------------------------------------------------------
__global__ void hist_gather_kernel(const float4* __restrict__ kc,
                                   const float4* __restrict__ vc,
                                   const int* __restrict__ bo,
                                   __bf16* __restrict__ Kb,
                                   __bf16* __restrict__ Vb) {
  int t = blockIdx.x * blockDim.x + threadIdx.x;  // B*H*HIST*(D/4)
  int d4 = t & 31;
  int idx = t >> 5;
  int pos = idx % HIST_;
  idx /= HIST_;
  int h = idx & (H_ - 1);
  int b = idx >> 5;

  int blk = bo[b * NBLK_ + (pos >> 6)];
  int slot = pos & (BS_ - 1);
  size_t src = (((size_t)blk * BS_ + slot) * H_ + h) * (D_ / 4) + d4;
  float4 kv = kc[src];
  float4 vv = vc[src];
  size_t dst = ((size_t)(b * H_ + h) * LKV_ + pos) * D_ + d4 * 4;
  Kb[dst + 0] = (__bf16)kv.x;
  Kb[dst + 1] = (__bf16)kv.y;
  Kb[dst + 2] = (__bf16)kv.z;
  Kb[dst + 3] = (__bf16)kv.w;
  Vb[dst + 0] = (__bf16)vv.x;
  Vb[dst + 1] = (__bf16)vv.y;
  Vb[dst + 2] = (__bf16)vv.z;
  Vb[dst + 3] = (__bf16)vv.w;
}

// ---------------------------------------------------------------------------
// Causal flash attention with WMMA. Grid (Lq/128, H, B); 8 waves/WG.
// Each wave owns 16 query rows; WG shares double-buffered K/V tiles
// (32 keys/step) in LDS. Next-step K is issued with
// global_load_async_to_lds_b128 (ASYNCcnt) before the current step's WMMAs,
// drained at the single end-of-step barrier. Q fragments load straight from
// global (rows are 16B-contiguous). Online softmax via half-wave __shfl_xor;
// P goes through wave-private LDS (s_wait_dscnt) to reshape C->A layout.
// ---------------------------------------------------------------------------
__global__ __launch_bounds__(256) void attn_kernel(
    const __bf16* __restrict__ Qb, const __bf16* __restrict__ Kb,
    const __bf16* __restrict__ Vb, __bf16* __restrict__ Ob) {
  __shared__ __align__(16) __bf16 sK[2][32 * 128];    // [key][d]
  __shared__ __align__(16) __bf16 sVt[2][128 * 32];   // [d][key] (transposed)
  __shared__ __align__(16) __bf16 sP[8 * 16 * 32];    // per-wave P staging

  const int tid = threadIdx.x;
  const int wid = tid >> 5;
  const int lane = tid & 31;
  const int lh = lane & 15;
  const bool hihalf = lane >= 16;
  const int qblk = blockIdx.x, h = blockIdx.y, b = blockIdx.z;
  const int qbase = qblk * 128;
  const size_t qoff = ((size_t)(b * H_ + h) * LQ_ + qbase) * D_;
  const size_t koff = (size_t)(b * H_ + h) * LKV_;

  // Q fragments straight from global: row m is 256B contiguous.
  const char* gQrow = reinterpret_cast<const char*>(Qb + qoff);
  v16bf qf[4];
#pragma unroll
  for (int kd = 0; kd < 4; ++kd) {
    int base = (wid * 16 + lh) * 256 + kd * 64 + (hihalf ? 16 : 0);
    qf[kd] = mem_frag(gQrow, base, base + 32);
  }

  const uint4* gK = reinterpret_cast<const uint4*>(Kb + koff * D_);

  // Stage key-block `kbase` into buffer `dst`.
  auto stageK = [&](int dst, int kbase) {
#pragma unroll
    for (int c = 0; c < 2; ++c) {
      int ch = tid + c * 256;
      int row = ch >> 4;
      int c8 = ch & 15;
#if USE_ASYNC
      __builtin_amdgcn_global_load_async_to_lds_b128(
          GPTR(gK + (size_t)(kbase + row) * 16 + c8),
          LPTR(&sK[dst][row * 128 + c8 * 8]), 0, 0);
#else
      *reinterpret_cast<uint4*>(&sK[dst][row * 128 + c8 * 8]) =
          gK[(size_t)(kbase + row) * 16 + c8];
#endif
    }
  };
  auto stageV = [&](int dst, int kbase) {
#pragma unroll
    for (int i = 0; i < 16; ++i) {
      int lin = tid + i * 256;
      int key = lin >> 7;
      int d = lin & 127;
      sVt[dst][d * 32 + key] = Vb[(koff + kbase + key) * D_ + d];
    }
  };

  // Prologue: stage block 0 into buffer 0.
  stageK(0, 0);
  stageV(0, 0);
#if USE_ASYNC
  __builtin_amdgcn_s_wait_asynccnt(0);
#endif
  __syncthreads();

  v8f acc[8] = {};
  float rowm[8], rowl[8];
#pragma unroll
  for (int r = 0; r < 8; ++r) { rowm[r] = -1e30f; rowl[r] = 0.0f; }

  const float scale = 0.08838834764831845f;  // 1/sqrt(128)
  const int kend = HIST_ + qbase + 128;      // causal bound for this block

  int cur = 0;
  for (int kb = 0; kb < kend; kb += 32) {
    const int nxt = cur ^ 1;
    const bool hasnext = (kb + 32 < kend);
    if (hasnext) stageK(nxt, kb + 32);  // async DMA overlaps current WMMAs

    // Scores S(16x32) = Q(16x128) . K^T: 8 WMMAs (4 d-chunks x 2 key-halves).
    const char* cK = reinterpret_cast<const char*>(sK[cur]);
    v8f s[2] = {};
#pragma unroll
    for (int kd = 0; kd < 4; ++kd) {
#pragma unroll
      for (int j = 0; j < 2; ++j) {
        int base = (j * 16 + lh) * 256 + kd * 64 + (hihalf ? 16 : 0);
        v16bf kf = mem_frag(cK, base, base + 32);
        s[j] = wmma_bf16(qf[kd], kf, s[j]);
      }
    }

    if (hasnext) stageV(nxt, kb + 32);  // sync V transpose into next buffer

    // Online softmax per row r (row M = r + 8*hihalf of this wave's tile).
#pragma unroll
    for (int r = 0; r < 8; ++r) {
      int m = r + (hihalf ? 8 : 0);
      int qpos = HIST_ + qbase + wid * 16 + m;
      float s0 = s[0][r] * scale;
      float s1 = s[1][r] * scale;
      if (kb + lh > qpos) s0 = -1e30f;
      if (kb + 16 + lh > qpos) s1 = -1e30f;
      float mloc = fmaxf(s0, s1);
#pragma unroll
      for (int x = 8; x >= 1; x >>= 1)
        mloc = fmaxf(mloc, __shfl_xor(mloc, x, 32));
      float mnew = fmaxf(rowm[r], mloc);
      float aold = __expf(rowm[r] - mnew);
      float p0 = __expf(s0 - mnew);
      float p1 = __expf(s1 - mnew);
      float lloc = p0 + p1;
#pragma unroll
      for (int x = 8; x >= 1; x >>= 1) lloc += __shfl_xor(lloc, x, 32);
      rowl[r] = rowl[r] * aold + lloc;
      rowm[r] = mnew;
#pragma unroll
      for (int dt = 0; dt < 8; ++dt) acc[dt][r] *= aold;
      // Stage P (bf16) into wave-private LDS region in [m][key] order.
      sP[wid * 512 + m * 32 + lh] = (__bf16)p0;
      sP[wid * 512 + m * 32 + 16 + lh] = (__bf16)p1;
    }
    // Wave-private LDS RAW: drain DS counter before reading P fragments.
    asm volatile("s_wait_dscnt 0x0" ::: "memory");

    const char* cP = reinterpret_cast<const char*>(sP);
    int basep = wid * 1024 + lh * 64 + (hihalf ? 16 : 0);
    v16bf pf = mem_frag(cP, basep, basep + 32);

    const char* cV = reinterpret_cast<const char*>(sVt[cur]);
#pragma unroll
    for (int dt = 0; dt < 8; ++dt) {
      int base = (dt * 16 + lh) * 64 + (hihalf ? 16 : 0);
      v16bf vf = mem_frag(cV, base, base + 32);
      acc[dt] = wmma_bf16(pf, vf, acc[dt]);
    }

#if USE_ASYNC
    __builtin_amdgcn_s_wait_asynccnt(0);  // next K tile landed
#endif
    __syncthreads();  // next buffers ready; cur free for reuse
    cur = nxt;
  }

  // Epilogue: O = acc / l, write [B][Lq][H][D] (== reshape(B*Lq, H*D)).
#pragma unroll
  for (int dt = 0; dt < 8; ++dt) {
#pragma unroll
    for (int r = 0; r < 8; ++r) {
      int m = r + (hihalf ? 8 : 0);
      int q = qbase + wid * 16 + m;
      int d = dt * 16 + lh;
      float v = acc[dt][r] / rowl[r];
      Ob[((size_t)(b * LQ_ + q) * H_ + h) * D_ + d] = (__bf16)v;
    }
  }
}

// ---------------------------------------------------------------------------
// Launch pipeline.
// ---------------------------------------------------------------------------
extern "C" void kernel_launch(void* const* d_in, const int* in_sizes, int n_in,
                              void* d_out, int out_size, void* d_ws,
                              size_t ws_size, hipStream_t stream) {
  (void)in_sizes; (void)n_in; (void)out_size; (void)ws_size;
  const float* hs = (const float*)d_in[0];
  const float* w_pack = (const float*)d_in[1];
  const float* w_o = (const float*)d_in[2];
  const float* k_cache = (const float*)d_in[3];
  const float* v_cache = (const float*)d_in[4];
  const int* block_offsets = (const int*)d_in[5];
  // d_in[6..8] = history_len / q_len / block_size (fixed-shape, hardcoded).

  char* ws = (char*)d_ws;
  size_t off = 0;
  auto alloc = [&](size_t bytes) -> void* {
    void* p = ws + off;
    off += (bytes + 255) & ~(size_t)255;
    return p;
  };
  __bf16* hs_bf   = (__bf16*)alloc((size_t)2048 * 4096 * 2);
  __bf16* wp_bf   = (__bf16*)alloc((size_t)12288 * 4096 * 2);
  __bf16* wo_bf   = (__bf16*)alloc((size_t)4096 * 4096 * 2);
  __bf16* qkv_bf  = (__bf16*)alloc((size_t)2048 * 12288 * 2);
  __bf16* q_bf    = (__bf16*)alloc((size_t)B_ * H_ * LQ_ * D_ * 2);
  __bf16* k_bf    = (__bf16*)alloc((size_t)B_ * H_ * LKV_ * D_ * 2);
  __bf16* v_bf    = (__bf16*)alloc((size_t)B_ * H_ * LKV_ * D_ * 2);
  __bf16* attn_bf = (__bf16*)alloc((size_t)2048 * 4096 * 2);

  cvt_f32_bf16<<<(2048 * 4096 / 4 + 255) / 256, 256, 0, stream>>>(
      (const float4*)hs, hs_bf, 2048 * 4096 / 4);
  cvt_f32_bf16<<<(12288 * 4096 / 4 + 255) / 256, 256, 0, stream>>>(
      (const float4*)w_pack, wp_bf, 12288 * 4096 / 4);
  cvt_f32_bf16<<<(4096 * 4096 / 4 + 255) / 256, 256, 0, stream>>>(
      (const float4*)w_o, wo_bf, 4096 * 4096 / 4);

  // QKV projection: (2048 x 4096) @ (12288 x 4096)^T -> bf16.
  gemm_bf16_wmma<<<dim3(12288 / 128, 2048 / 128), 256, 0, stream>>>(
      hs_bf, wp_bf, qkv_bf, 2048, 12288, 4096, 1);

  rope_split_kernel<<<(B_ * LQ_ * H_ * 64) / 256, 256, 0, stream>>>(
      qkv_bf, q_bf, k_bf, v_bf);

  hist_gather_kernel<<<(B_ * H_ * HIST_ * (D_ / 4)) / 256, 256, 0, stream>>>(
      (const float4*)k_cache, (const float4*)v_cache, block_offsets, k_bf, v_bf);

  attn_kernel<<<dim3(LQ_ / 128, H_, B_), 256, 0, stream>>>(q_bf, k_bf, v_bf,
                                                           attn_bf);

  // Output projection: (2048 x 4096) @ (4096 x 4096)^T -> fp32 d_out.
  gemm_bf16_wmma<<<dim3(4096 / 128, 2048 / 128), 256, 0, stream>>>(
      attn_bf, wo_bf, d_out, 2048, 4096, 4096, 0);
}
